// LaneATT2DHead_45028437131812
// MI455X (gfx1250) — matrix operand order repaired and checked by reading, not compiled, not run on hardware
//
#include <hip/hip_runtime.h>
#include <hip/hip_bf16.h>
#include <math.h>

// ---------------------------------------------------------------------------
// LaneATT head for MI455X (gfx1250, wave32).
// Big GEMMs (scores: 22272x1408x2783, attention: 8x[2784x2784x1408], heads:
// 22272x2816x76) run on V_WMMA_F32_16X16X32_BF16 with fp32 accumulation.
// Everything bandwidth-bound stays fp32 VALU. ~360 GFLOP total, ~0.5 GB HBM
// traffic -> compute-bound on the WMMA pipe; bf16 is the 8x-throughput path
// vs the f32 16x16x4 WMMA.
// ---------------------------------------------------------------------------

#define BATCH   8
#define PANC    2784          // number of anchors (16*174 = 32*87)
#define PM1     2783
#define FH      22
#define FW      40
#define AFC     64
#define INCH    256
#define DFEAT   1408          // AFC*FH = 32*44
#define BP      (BATCH*PANC)  // 22272 = 64*348
#define K2      2816          // 2*DFEAT = 32*88
#define HEADN   76            // 2 cls + 1 so + 73 reg
#define HEADNP  80            // padded to 5 N-tiles of 16
#define OUTW    77            // reg_proposals row width
#define LP      48            // LDS tile pitch in u16 elements (96B, 16B-aligned)

typedef unsigned short u16;
typedef __attribute__((ext_vector_type(16))) __bf16 v16bf;
typedef __attribute__((ext_vector_type(8)))  float  v8f;

// fp32 -> bf16, round-to-nearest-even
__device__ __forceinline__ u16 f2bf(float f) {
  union { float f; unsigned u; } v; v.f = f;
  unsigned r = v.u + 0x7FFFu + ((v.u >> 16) & 1u);
  return (u16)(r >> 16);
}

// A-matrix fragment (16x32 bf16). Per ISA 7.12.2: lanes 0-15 hold M=lane,
// K={0..7,16..23}; lanes 16-31 hold M=lane-16, K={8..15,24..31}.
__device__ __forceinline__ v16bf frag_a(const u16* tile, int lane) {
  const int row = lane & 15, half = lane >> 4;
  union { uint4 u[2]; v16bf v; } r;
  const u16* p = tile + row * LP + half * 8;
  r.u[0] = *(const uint4*)(p);        // K half*8 .. half*8+7
  r.u[1] = *(const uint4*)(p + 16);   // K 16+half*8 .. 16+half*8+7
  return r.v;
}

// B-matrix fragment (32x16 bf16), tile stored N-major (row = N, K contiguous).
// Same K order as frag_a: lanes 0-15 hold N=lane, K={0..7,16..23}; lanes 16-31
// hold N=lane-16, K={8..15,24..31}.
__device__ __forceinline__ v16bf frag_b(const u16* tile, int lane) {
  const int row = lane & 15, half = lane >> 4;
  union { uint4 u[2]; v16bf v; } r;
  const u16* p = tile + row * LP + half * 8;
  r.u[0] = *(const uint4*)(p);
  r.u[1] = *(const uint4*)(p + 16);
  return r.v;
}

__device__ __forceinline__ v8f wmma_bf16(v16bf a, v16bf b, v8f c) {
  return __builtin_amdgcn_wmma_f32_16x16x32_bf16(false, a, false, b, (short)0, c,
                                                 false, false);
}

// ---------------------------------------------------------------------------
// 1) 1x1 conv: feat[b][o][h][w] = sum_c x[b][c][h][w] * conv_w[o][c] + b[o]
// ---------------------------------------------------------------------------
__global__ void __launch_bounds__(256)
conv1x1_kernel(const float* __restrict__ x, const float* __restrict__ w,
               const float* __restrict__ bias, float* __restrict__ feat) {
  int idx = blockIdx.x * 256 + threadIdx.x;           // over BATCH*AFC*FH*FW
  if (idx >= BATCH * AFC * FH * FW) return;
  int hw = idx % (FH * FW);
  int o  = (idx / (FH * FW)) % AFC;
  int b  = idx / (FH * FW * AFC);
  const float* xp = x + (size_t)b * INCH * FH * FW + hw;
  const float* wp = w + o * INCH;
  float acc = bias[o];
#pragma unroll 8
  for (int c = 0; c < INCH; ++c) acc = fmaf(xp[(size_t)c * FH * FW], wp[c], acc);
  feat[idx] = acc;
}

// ---------------------------------------------------------------------------
// 2) anchor gather -> baf (bf16): baf[b][p][a*FH+h] = feat[b][a][h][cut_xs[p][h]]
// ---------------------------------------------------------------------------
__global__ void __launch_bounds__(256)
gather_kernel(const float* __restrict__ feat, const int* __restrict__ cut_xs,
              const unsigned char* __restrict__ invalid, u16* __restrict__ baf) {
  long long idx = (long long)blockIdx.x * 256 + threadIdx.x;   // BP*DFEAT
  if (idx >= (long long)BP * DFEAT) return;
  int d = (int)(idx % DFEAT);
  int p = (int)((idx / DFEAT) % PANC);
  int b = (int)(idx / ((long long)DFEAT * PANC));
  int a = d / FH, h = d % FH;
  float v = 0.0f;
  if (!invalid[p * FH + h]) {                         // numpy bool -> 1 byte
    int xc = cut_xs[p * FH + h];
    v = feat[(((size_t)b * AFC + a) * FH + h) * FW + xc];
  }
  baf[idx] = f2bf(v);
}

// ---------------------------------------------------------------------------
// 3) weight conversions
// ---------------------------------------------------------------------------
__global__ void __launch_bounds__(256)
cvt_attw_kernel(const float* __restrict__ attw, u16* __restrict__ out) {
  int idx = blockIdx.x * 256 + threadIdx.x;
  if (idx < PM1 * DFEAT) out[idx] = f2bf(attw[idx]);
}

__global__ void __launch_bounds__(256)
build_head_kernel(const float* __restrict__ cls_w, const float* __restrict__ so_w,
                  const float* __restrict__ reg_w, const float* __restrict__ cls_b,
                  const float* __restrict__ so_b, const float* __restrict__ reg_b,
                  u16* __restrict__ wc, float* __restrict__ bias76) {
  int idx = blockIdx.x * 256 + threadIdx.x;           // HEADNP*K2
  if (idx >= HEADNP * K2) return;
  int o = idx / K2, k = idx % K2;
  float v = 0.0f;
  if (o < 2)           v = cls_w[o * K2 + k];
  else if (o == 2)     v = so_w[k];
  else if (o < HEADN)  v = reg_w[(o - 3) * K2 + k];
  wc[idx] = f2bf(v);
  if (k == 0 && o < HEADN)
    bias76[o] = (o < 2) ? cls_b[o] : (o == 2) ? so_b[0] : reg_b[o - 3];
}

// ---------------------------------------------------------------------------
// 4) scores GEMM: S[row][j] = baf[row] . att_w[j] + att_b[j]
//    scattered straight into M with column shift q = j + (j >= p).
//    Block tile 64x64, 4 waves, K = 1408 (44 steps).
// ---------------------------------------------------------------------------
__global__ void __launch_bounds__(128)
scores_gemm_kernel(const u16* __restrict__ baf, const u16* __restrict__ attw,
                   const float* __restrict__ att_b, float* __restrict__ Mout) {
  __shared__ __align__(16) u16 As[64 * LP];
  __shared__ __align__(16) u16 Bs[64 * LP];
  const int tid = threadIdx.x, lane = tid & 31, wave = tid >> 5;
  const int row0 = blockIdx.x * 64;   // global BP row (batch-agnostic: shared B)
  const int col0 = blockIdx.y * 64;   // j in [0, 2816)
  v8f acc[4];
#pragma unroll
  for (int n = 0; n < 4; ++n) acc[n] = (v8f){0, 0, 0, 0, 0, 0, 0, 0};

  for (int kt = 0; kt < DFEAT / 32; ++kt) {
    // A: 64 rows x 32 K, 16B vector loads, 2 per thread
#pragma unroll
    for (int i = tid; i < 256; i += 128) {
      int r = i >> 2, c = (i & 3) * 8;
      *(uint4*)(As + r * LP + c) =
          *(const uint4*)(baf + (size_t)(row0 + r) * DFEAT + kt * 32 + c);
    }
    // B: 64 att_w rows (N-major, K contiguous), guard n < 2783
#pragma unroll
    for (int i = tid; i < 256; i += 128) {
      int r = i >> 2, c = (i & 3) * 8;
      uint4 v = {0, 0, 0, 0};
      int n = col0 + r;
      if (n < PM1) v = *(const uint4*)(attw + (size_t)n * DFEAT + kt * 32 + c);
      *(uint4*)(Bs + r * LP + c) = v;
    }
    __syncthreads();
    v16bf a = frag_a(As + wave * 16 * LP, lane);
#pragma unroll
    for (int nt = 0; nt < 4; ++nt) {
      v16bf b = frag_b(Bs + nt * 16 * LP, lane);
      acc[nt] = wmma_bf16(a, b, acc[nt]);
    }
    __syncthreads();
  }
  // epilogue: C layout -> m = i + 8*(lane>=16), n = lane&15
  const int nloc = lane & 15, mhi = (lane >> 4) * 8;
#pragma unroll
  for (int nt = 0; nt < 4; ++nt) {
#pragma unroll
    for (int i = 0; i < 8; ++i) {
      int row = row0 + wave * 16 + mhi + i;
      int j = col0 + nt * 16 + nloc;
      if (j < PM1) {
        int b = row / PANC, p = row % PANC;
        int q = j + (j >= p);                         // skip diagonal
        Mout[(size_t)b * PANC * PANC + (size_t)p * PANC + q] = acc[nt][i] + att_b[j];
      }
    }
  }
}

// ---------------------------------------------------------------------------
// 5) in-place row softmax over M (length P, diagonal excluded then set to 0)
// ---------------------------------------------------------------------------
__global__ void __launch_bounds__(256)
softmax_kernel(float* __restrict__ Mout) {
  const int row = blockIdx.x;                         // 0..BP-1
  const int b = row / PANC, p = row % PANC;
  float* r = Mout + (size_t)b * PANC * PANC + (size_t)p * PANC;
  __shared__ float red[256];
  float mx = -1e30f;
  for (int q = threadIdx.x; q < PANC; q += 256)
    if (q != p) mx = fmaxf(mx, r[q]);
  red[threadIdx.x] = mx;
  __syncthreads();
  for (int s = 128; s > 0; s >>= 1) {
    if (threadIdx.x < s) red[threadIdx.x] = fmaxf(red[threadIdx.x], red[threadIdx.x + s]);
    __syncthreads();
  }
  mx = red[0];
  __syncthreads();
  float sm = 0.0f;
  for (int q = threadIdx.x; q < PANC; q += 256)
    if (q != p) sm += __expf(r[q] - mx);
  red[threadIdx.x] = sm;
  __syncthreads();
  for (int s = 128; s > 0; s >>= 1) {
    if (threadIdx.x < s) red[threadIdx.x] += red[threadIdx.x + s];
    __syncthreads();
  }
  const float inv = 1.0f / red[0];
  for (int q = threadIdx.x; q < PANC; q += 256)
    r[q] = (q == p) ? 0.0f : __expf(r[q] - mx) * inv;
}

// ---------------------------------------------------------------------------
// 6) attention GEMM (per batch): att[b][p][d] = sum_q M[b][p][q]*baf[b][q][d]
//    Block tile 32x64, 2 waves, K = P (87 steps). A staged f32->bf16,
//    B staged transposed (N-major) into LDS.
// ---------------------------------------------------------------------------
__global__ void __launch_bounds__(64)
att_gemm_kernel(const float* __restrict__ Mmat, const u16* __restrict__ baf,
                u16* __restrict__ attf) {
  __shared__ __align__(16) u16 As[32 * LP];
  __shared__ __align__(16) u16 Bs[64 * LP];
  const int tid = threadIdx.x, lane = tid & 31, wave = tid >> 5;
  const int b = blockIdx.z;
  const int p0 = blockIdx.x * 32, d0 = blockIdx.y * 64;
  const float* Mb = Mmat + (size_t)b * PANC * PANC;
  const u16*   Bg = baf + (size_t)b * PANC * DFEAT;
  v8f acc[4];
#pragma unroll
  for (int n = 0; n < 4; ++n) acc[n] = (v8f){0, 0, 0, 0, 0, 0, 0, 0};

  for (int kt = 0; kt < PANC / 32; ++kt) {
    const int q0 = kt * 32;
    // A: 32 rows(p) x 32 cols(q) f32 -> bf16; 256 float4, 4 per thread
#pragma unroll
    for (int i = tid; i < 256; i += 64) {
      int r = i >> 3, c = (i & 7) * 4;
      float4 v = *(const float4*)(Mb + (size_t)(p0 + r) * PANC + q0 + c);
      ushort4 s;
      s.x = f2bf(v.x); s.y = f2bf(v.y); s.z = f2bf(v.z); s.w = f2bf(v.w);
      *(ushort4*)(As + r * LP + c) = s;
    }
    // B: Bs[n][k] = baf[b][q0+k][d0+n]; read 8 contiguous n per k, scatter
#pragma unroll
    for (int i = tid; i < 256; i += 64) {
      int k = i >> 3, n8 = (i & 7) * 8;
      union { uint4 u; u16 e[8]; } v;
      v.u = *(const uint4*)(Bg + (size_t)(q0 + k) * DFEAT + d0 + n8);
#pragma unroll
      for (int e = 0; e < 8; ++e) Bs[(n8 + e) * LP + k] = v.e[e];
    }
    __syncthreads();
    v16bf a = frag_a(As + wave * 16 * LP, lane);
#pragma unroll
    for (int nt = 0; nt < 4; ++nt) {
      v16bf bb = frag_b(Bs + nt * 16 * LP, lane);
      acc[nt] = wmma_bf16(a, bb, acc[nt]);
    }
    __syncthreads();
  }
  const int nloc = lane & 15, mhi = (lane >> 4) * 8;
#pragma unroll
  for (int nt = 0; nt < 4; ++nt) {
#pragma unroll
    for (int i = 0; i < 8; ++i) {
      int p = p0 + wave * 16 + mhi + i;
      int d = d0 + nt * 16 + nloc;
      attf[((size_t)b * PANC + p) * DFEAT + d] = f2bf(acc[nt][i]);
    }
  }
}

// ---------------------------------------------------------------------------
// 7) head GEMM over feats = [att_feats | baf] (K=2816), N=80 (76 used),
//    fused epilogue: bias + sigmoid(so) + anchor adds -> reg_proposals.
// ---------------------------------------------------------------------------
__global__ void __launch_bounds__(128)
head_gemm_kernel(const u16* __restrict__ attf, const u16* __restrict__ baf,
                 const u16* __restrict__ wc, const float* __restrict__ bias76,
                 const float* __restrict__ anchors, float* __restrict__ out) {
  __shared__ __align__(16) u16 As[64 * LP];
  __shared__ __align__(16) u16 Bs[HEADNP * LP];
  const int tid = threadIdx.x, lane = tid & 31, wave = tid >> 5;
  const int row0 = blockIdx.x * 64;
  v8f acc[5];
#pragma unroll
  for (int n = 0; n < 5; ++n) acc[n] = (v8f){0, 0, 0, 0, 0, 0, 0, 0};

  for (int kt = 0; kt < K2 / 32; ++kt) {
    const u16* Ag = (kt < DFEAT / 32) ? attf : baf;
    const int  kc = (kt < DFEAT / 32) ? kt * 32 : (kt - DFEAT / 32) * 32;
#pragma unroll
    for (int i = tid; i < 256; i += 128) {
      int r = i >> 2, c = (i & 3) * 8;
      *(uint4*)(As + r * LP + c) =
          *(const uint4*)(Ag + (size_t)(row0 + r) * DFEAT + kc + c);
    }
    for (int i = tid; i < 320; i += 128) {            // 80 rows x 4 chunks
      int r = i >> 2, c = (i & 3) * 8;
      *(uint4*)(Bs + r * LP + c) = *(const uint4*)(wc + (size_t)r * K2 + kt * 32 + c);
    }
    __syncthreads();
    v16bf a = frag_a(As + wave * 16 * LP, lane);
#pragma unroll
    for (int nt = 0; nt < 5; ++nt) {
      v16bf bb = frag_b(Bs + nt * 16 * LP, lane);
      acc[nt] = wmma_bf16(a, bb, acc[nt]);
    }
    __syncthreads();
  }
  const int nloc = lane & 15, mhi = (lane >> 4) * 8;
#pragma unroll
  for (int nt = 0; nt < 5; ++nt) {
    int o = nt * 16 + nloc;
    if (o >= HEADN) continue;
#pragma unroll
    for (int i = 0; i < 8; ++i) {
      int row = row0 + wave * 16 + mhi + i;
      int b = row / PANC, p = row % PANC;
      float v = acc[nt][i] + bias76[o];
      float* op = out + ((size_t)b * PANC + p) * OUTW;
      const float* anc = anchors + (size_t)p * OUTW;
      if (o < 2) {
        op[o] = v;                                    // cls logits
      } else if (o == 2) {
        op[2] = (anc[2] + 1.0f / (1.0f + __expf(-v))) * 0.5f;
        op[3] = anc[3];                               // pass-through column
      } else {
        op[4 + (o - 3)] = v + anc[4 + (o - 3)];       // anchors + reg
      }
    }
  }
}

// ---------------------------------------------------------------------------
extern "C" void kernel_launch(void* const* d_in, const int* in_sizes, int n_in,
                              void* d_out, int out_size, void* d_ws, size_t ws_size,
                              hipStream_t stream) {
  (void)in_sizes; (void)n_in; (void)out_size; (void)ws_size;
  const float* x       = (const float*)d_in[0];
  const float* conv_w  = (const float*)d_in[1];
  const float* conv_b  = (const float*)d_in[2];
  const float* att_w   = (const float*)d_in[3];
  const float* att_b   = (const float*)d_in[4];
  const float* cls_w   = (const float*)d_in[5];
  const float* cls_b   = (const float*)d_in[6];
  const float* so_w    = (const float*)d_in[7];
  const float* so_b    = (const float*)d_in[8];
  const float* reg_w   = (const float*)d_in[9];
  const float* reg_b   = (const float*)d_in[10];
  const float* anchors = (const float*)d_in[11];
  const int*   cut_xs  = (const int*)d_in[12];
  const unsigned char* invalid = (const unsigned char*)d_in[13]; // numpy bool

  float* out      = (float*)d_out;
  float* out_reg  = out;                                   // BATCH*P*77
  float* out_M    = out + (size_t)BATCH * PANC * OUTW;     // BATCH*P*P
  float* out_feat = out_M + (size_t)BATCH * PANC * PANC;   // BATCH*AFC*FH*FW

  // workspace carve-up (~134 MB)
  char* ws = (char*)d_ws;
  size_t off = 0;
  auto take = [&](size_t bytes) {
    char* pp = ws + off;
    off = (off + bytes + 255) & ~(size_t)255;
    return pp;
  };
  u16*   baf    = (u16*)take((size_t)BP * DFEAT * 2);      // 62.7 MB
  u16*   attf   = (u16*)take((size_t)BP * DFEAT * 2);      // 62.7 MB
  u16*   attw16 = (u16*)take((size_t)PM1 * DFEAT * 2);     //  7.8 MB
  u16*   wc16   = (u16*)take((size_t)HEADNP * K2 * 2);     //  0.45 MB
  float* bias76 = (float*)take(HEADN * sizeof(float));

  conv1x1_kernel<<<dim3((BATCH * AFC * FH * FW) / 256), dim3(256), 0, stream>>>(
      x, conv_w, conv_b, out_feat);
  gather_kernel<<<dim3((unsigned)(((long long)BP * DFEAT) / 256)), dim3(256), 0,
                  stream>>>(out_feat, cut_xs, invalid, baf);
  cvt_attw_kernel<<<dim3((PM1 * DFEAT + 255) / 256), dim3(256), 0, stream>>>(
      att_w, attw16);
  build_head_kernel<<<dim3((HEADNP * K2) / 256), dim3(256), 0, stream>>>(
      cls_w, so_w, reg_w, cls_b, so_b, reg_b, wc16, bias76);
  scores_gemm_kernel<<<dim3(BP / 64, (PANC + 63) / 64), dim3(128), 0, stream>>>(
      baf, attw16, att_b, out_M);
  softmax_kernel<<<dim3(BP), dim3(256), 0, stream>>>(out_M);
  att_gemm_kernel<<<dim3(PANC / 32, DFEAT / 64, BATCH), dim3(64), 0, stream>>>(
      out_M, baf, attf);
  head_gemm_kernel<<<dim3(BP / 64), dim3(128), 0, stream>>>(
      attf, baf, wc16, bias76, anchors, out_reg);
}
